// InferenceModule_51969104281852
// MI455X (gfx1250) — matrix operand order repaired and checked by reading, not compiled, not run on hardware
//
#include <hip/hip_runtime.h>
#include <hip/hip_bf16.h>
#include <cstdint>
#include <cstddef>

// ---------------- problem constants (from reference) ----------------
#define BB   64
#define NN   196
#define EE   2048
#define DD   512
#define AA   512
#define EMB  512
#define VV   10000
#define TT   20
#define VPAD 10048          // 10000 rounded up to multiple of 64
#define KCAT 3072           // EM + E + D  (inp | awe | h)

typedef __attribute__((ext_vector_type(16))) __bf16 v16bf;
typedef __attribute__((ext_vector_type(8)))  float  v8f;

union FragBF16 {
    v16bf v;
    uint4 q[2];
};

__device__ __forceinline__ float sigmoidf_(float x) { return 1.0f / (1.0f + __expf(-x)); }

// ---------------- elementwise f32 -> bf16 ----------------
__global__ void f2bf_kernel(const float* __restrict__ in, __hip_bfloat16* __restrict__ out, size_t n) {
    size_t i = (size_t)blockIdx.x * blockDim.x + threadIdx.x;
    if (i < n) out[i] = __float2bfloat16(in[i]);
}

// ---------------- transpose + convert: in[K, Nreal] f32 -> out[Npad, K] bf16 (zero pad) ----------------
__global__ void transpose_f2bf_kernel(const float* __restrict__ in, __hip_bfloat16* __restrict__ out,
                                      int K, int Nreal, int Npad) {
    size_t i = (size_t)blockIdx.x * blockDim.x + threadIdx.x;
    size_t total = (size_t)Npad * K;
    if (i >= total) return;
    int n = (int)(i / K);
    int k = (int)(i % K);
    float v = (n < Nreal) ? in[(size_t)k * Nreal + n] : 0.0f;
    out[i] = __float2bfloat16(v);
}

// ---------------- build transposed fused LSTM weight: out[2048, 3072] ----------------
__global__ void build_wcat_kernel(const float* __restrict__ Wih,  // [2560, 2048]
                                  const float* __restrict__ Whh,  // [512, 2048]
                                  __hip_bfloat16* __restrict__ out) {
    size_t i = (size_t)blockIdx.x * blockDim.x + threadIdx.x;
    size_t total = (size_t)(4 * DD) * KCAT;
    if (i >= total) return;
    int n = (int)(i / KCAT);
    int k = (int)(i % KCAT);
    float v = (k < EMB + EE) ? Wih[(size_t)k * (4 * DD) + n]
                             : Whh[(size_t)(k - (EMB + EE)) * (4 * DD) + n];
    out[i] = __float2bfloat16(v);
}

// ---------------- mean over N axis: feat[B,N,E] -> mean_bf[B,E] ----------------
__global__ void mean_enc_kernel(const float* __restrict__ feat, __hip_bfloat16* __restrict__ mean_bf) {
    int idx = blockIdx.x * blockDim.x + threadIdx.x;   // over B*E
    if (idx >= BB * EE) return;
    int b = idx / EE, e = idx % EE;
    const float* p = feat + (size_t)b * NN * EE + e;
    float s = 0.0f;
    for (int n = 0; n < NN; ++n) s += p[(size_t)n * EE];
    mean_bf[idx] = __float2bfloat16(s * (1.0f / NN));
}

// ---------------- init inp = embedding[1] broadcast ----------------
__global__ void init_inp_kernel(const float* __restrict__ embedding, __hip_bfloat16* __restrict__ inp_bf) {
    int idx = blockIdx.x * blockDim.x + threadIdx.x;   // over B*EM
    if (idx >= BB * EMB) return;
    int e = idx % EMB;
    inp_bf[idx] = __float2bfloat16(embedding[(size_t)1 * EMB + e]);
}

// ---------------- WMMA bf16 GEMM: C[M,Nld] = act(A[M,K] @ Bt[Nld,K]^T + bias) ----------------
// Block: 256 threads (8 wave32 waves). Block tile 64(M) x 64(N).
// Wave w: m sub-tile = (w&3)*16, n sub-tile = (w>>2)*32 (two 16x16 WMMA accum tiles).
template <int ACT>   // 0 = none, 1 = sigmoid
__global__ void __launch_bounds__(256)
gemm_bf16_kernel(const __hip_bfloat16* __restrict__ A,    // [M, K]
                 const __hip_bfloat16* __restrict__ Bt,   // [Nld, K] (transposed weights)
                 const float* __restrict__ bias, int biasN,
                 float* __restrict__ C,                   // optional [M, Nld]
                 __hip_bfloat16* __restrict__ Cb,         // optional [M, Nld]
                 int M, int Nld, int K) {
    const int tid  = threadIdx.x;
    const int lane = tid & 31;
    const int wave = tid >> 5;
    const int wm   = wave & 3;          // 0..3  -> +16 rows each
    const int wn   = wave >> 2;         // 0..1  -> +32 cols each
    const int mBlock = blockIdx.y * 64;
    const int nBlock = blockIdx.x * 64;

    // A fragment addressing: lane holds row m = (lane&15); 16 contiguous K starting at (lane>>4)*16.
    const int mRowA = mBlock + wm * 16 + (lane & 15);
    const int aK    = (lane >> 4) * 16;
    // B fragment addressing: lane holds col n = (lane&15); 16 contiguous K starting at (lane>>4)*16.
    const int n0 = nBlock + wn * 32 + (lane & 15);
    const int n1 = n0 + 16;
    const int bK = (lane >> 4) * 16;

    const __hip_bfloat16* aRow  = A  + (size_t)mRowA * K;
    const __hip_bfloat16* bRow0 = Bt + (size_t)n0 * K;
    const __hip_bfloat16* bRow1 = Bt + (size_t)n1 * K;

    v8f acc0 = {0.f, 0.f, 0.f, 0.f, 0.f, 0.f, 0.f, 0.f};
    v8f acc1 = {0.f, 0.f, 0.f, 0.f, 0.f, 0.f, 0.f, 0.f};

    for (int k0 = 0; k0 < K; k0 += 32) {
        FragBF16 fa, fb0, fb1;
        fa.q[0]  = *reinterpret_cast<const uint4*>(aRow  + k0 + aK);
        fa.q[1]  = *reinterpret_cast<const uint4*>(aRow  + k0 + aK + 8);
        fb0.q[0] = *reinterpret_cast<const uint4*>(bRow0 + k0 + bK);
        fb0.q[1] = *reinterpret_cast<const uint4*>(bRow0 + k0 + bK + 8);
        fb1.q[0] = *reinterpret_cast<const uint4*>(bRow1 + k0 + bK);
        fb1.q[1] = *reinterpret_cast<const uint4*>(bRow1 + k0 + bK + 8);
        // prefetch next K tile (speculative; OOB dropped)
        __builtin_prefetch(aRow  + k0 + 32 + aK, 0, 1);
        __builtin_prefetch(bRow0 + k0 + 32 + bK, 0, 1);
        __builtin_prefetch(bRow1 + k0 + 32 + bK, 0, 1);

        acc0 = __builtin_amdgcn_wmma_f32_16x16x32_bf16(false, fa.v, false, fb0.v,
                                                       (short)0, acc0, false, false);
        acc1 = __builtin_amdgcn_wmma_f32_16x16x32_bf16(false, fa.v, false, fb1.v,
                                                       (short)0, acc1, false, false);
    }

    // C/D layout: VGPR r -> row = r + 8*(lane>>4); col = lane&15 within tile.
    const int mTop = mBlock + wm * 16 + 8 * (lane >> 4);
#pragma unroll
    for (int r = 0; r < 8; ++r) {
        int m = mTop + r;
        float v0 = acc0[r] + ((bias && n0 < biasN) ? bias[n0] : 0.0f);
        float v1 = acc1[r] + ((bias && n1 < biasN) ? bias[n1] : 0.0f);
        if (ACT == 1) { v0 = sigmoidf_(v0); v1 = sigmoidf_(v1); }
        size_t i0 = (size_t)m * Nld + n0;
        size_t i1 = (size_t)m * Nld + n1;
        if (C)  { C[i0]  = v0; C[i1] = v1; }
        if (Cb) { Cb[i0] = __float2bfloat16(v0); Cb[i1] = __float2bfloat16(v1); }
    }
}

// ---------------- attention: scores -> softmax -> gated weighted sum into x[:, 512:2560] ----------------
__global__ void __launch_bounds__(256)
attention_kernel(const float* __restrict__ att1,    // [B, N, A]
                 const float* __restrict__ att2,    // [B, A]
                 const float* __restrict__ wfull,   // [A]
                 const float* __restrict__ enc,     // [B, N, E]
                 const float* __restrict__ gate,    // [B, E]
                 __hip_bfloat16* __restrict__ x_bf) // [B, KCAT]
{
    const int b   = blockIdx.x;
    const int tid = threadIdx.x;
    __shared__ float sc[256];
    __shared__ float red[256];

    float score = -1e30f;
    if (tid < NN) {
        const float* a1 = att1 + ((size_t)b * NN + tid) * AA;
        const float* a2 = att2 + (size_t)b * AA;
        float s = 0.0f;
        for (int a = 0; a < AA; ++a) {
            float t = a1[a] + a2[a];
            t = t > 0.0f ? t : 0.0f;
            s += t * wfull[a];
        }
        score = s;
    }
    sc[tid] = score;
    red[tid] = score;
    __syncthreads();
    for (int s = 128; s > 0; s >>= 1) {
        if (tid < s) red[tid] = fmaxf(red[tid], red[tid + s]);
        __syncthreads();
    }
    float mx = red[0];
    __syncthreads();
    float ex = (tid < NN) ? __expf(sc[tid] - mx) : 0.0f;
    red[tid] = ex;
    __syncthreads();
    for (int s = 128; s > 0; s >>= 1) {
        if (tid < s) red[tid] += red[tid + s];
        __syncthreads();
    }
    float inv = 1.0f / red[0];
    __syncthreads();
    sc[tid] = ex * inv;          // alpha
    __syncthreads();

    const float* encb = enc + (size_t)b * NN * EE;
    for (int e = tid; e < EE; e += 256) {
        float s = 0.0f;
        for (int n = 0; n < NN; ++n) s += sc[n] * encb[(size_t)n * EE + e];
        float g = gate[(size_t)b * EE + e];
        x_bf[(size_t)b * KCAT + EMB + e] = __float2bfloat16(s * g);
    }
}

// ---------------- assemble x = [inp | (awe, already written) | h] ----------------
__global__ void build_x_kernel(const __hip_bfloat16* __restrict__ inp_bf,  // [B, EM]
                               const __hip_bfloat16* __restrict__ h_bf,    // [B, D]
                               __hip_bfloat16* __restrict__ x_bf) {        // [B, KCAT]
    int idx = blockIdx.x * blockDim.x + threadIdx.x;     // over B*(EM+D)
    if (idx >= BB * (EMB + DD)) return;
    int per = EMB + DD;
    int b = idx / per, j = idx % per;
    if (j < EMB) x_bf[(size_t)b * KCAT + j] = inp_bf[(size_t)b * EMB + j];
    else         x_bf[(size_t)b * KCAT + (EMB + EE) + (j - EMB)] = h_bf[(size_t)b * DD + (j - EMB)];
}

// ---------------- LSTM pointwise ----------------
__global__ void lstm_kernel(const float* __restrict__ z,      // [B, 4D] gates i,f,g,o
                            float* __restrict__ c,            // [B, D] inout
                            __hip_bfloat16* __restrict__ h_bf) {
    int idx = blockIdx.x * blockDim.x + threadIdx.x;          // over B*D
    if (idx >= BB * DD) return;
    int b = idx / DD, d = idx % DD;
    const float* zb = z + (size_t)b * (4 * DD);
    float zi = zb[d], zf = zb[DD + d], zg = zb[2 * DD + d], zo = zb[3 * DD + d];
    float cn = sigmoidf_(zf) * c[idx] + sigmoidf_(zi) * tanhf(zg);
    float hn = sigmoidf_(zo) * tanhf(cn);
    c[idx] = cn;
    h_bf[idx] = __float2bfloat16(hn);
}

// ---------------- argmax over vocab + embedding gather ----------------
__global__ void __launch_bounds__(256)
argmax_embed_kernel(const float* __restrict__ preds,      // [B, VPAD]
                    const float* __restrict__ embedding,  // [V, EM]
                    __hip_bfloat16* __restrict__ inp_bf,  // [B, EM]
                    int* __restrict__ ids, int step) {
    const int b = blockIdx.x, tid = threadIdx.x;
    __shared__ float smax[256];
    __shared__ int   sidx[256];
    float best = -1e30f; int bi = 0x7fffffff;
    const float* p = preds + (size_t)b * VPAD;
    for (int v = tid; v < VV; v += 256) {
        float val = p[v];
        if (val > best || (val == best && v < bi)) { best = val; bi = v; }
    }
    smax[tid] = best; sidx[tid] = bi;
    __syncthreads();
    for (int s = 128; s > 0; s >>= 1) {
        if (tid < s) {
            if (smax[tid + s] > smax[tid] ||
                (smax[tid + s] == smax[tid] && sidx[tid + s] < sidx[tid])) {
                smax[tid] = smax[tid + s]; sidx[tid] = sidx[tid + s];
            }
        }
        __syncthreads();
    }
    int pid = sidx[0];
    if (tid == 0) ids[b * TT + step] = pid;
    const float* er = embedding + (size_t)pid * EMB;
    for (int e = tid; e < EMB; e += 256) inp_bf[(size_t)b * EMB + e] = __float2bfloat16(er[e]);
}

// =====================================================================
extern "C" void kernel_launch(void* const* d_in, const int* in_sizes, int n_in,
                              void* d_out, int out_size, void* d_ws, size_t ws_size,
                              hipStream_t stream) {
    const float* features  = (const float*)d_in[0];   // [B,N,E]
    const float* embedding = (const float*)d_in[1];   // [V,EM]
    const float* W_enc_att = (const float*)d_in[2];   // [E,A]
    const float* W_dec_att = (const float*)d_in[3];   // [D,A]
    const float* w_full    = (const float*)d_in[4];   // [A]
    const float* W_init_h  = (const float*)d_in[5];   // [E,D]
    const float* b_init_h  = (const float*)d_in[6];   // [D]
    const float* W_init_c  = (const float*)d_in[7];   // [E,D]
    const float* b_init_c  = (const float*)d_in[8];   // [D]
    const float* W_fbeta   = (const float*)d_in[9];   // [D,E]
    const float* b_fbeta   = (const float*)d_in[10];  // [E]
    const float* W_ih      = (const float*)d_in[11];  // [2560,4D]
    const float* W_hh      = (const float*)d_in[12];  // [D,4D]
    const float* b_lstm    = (const float*)d_in[13];  // [4D]
    const float* W_fc      = (const float*)d_in[14];  // [D,V]
    const float* b_fc      = (const float*)d_in[15];  // [V]
    // d_in[16] = max_length (=20, hardcoded as TT)

    int* ids_out = (int*)d_out;

    // -------- workspace layout --------
    char* ws = (char*)d_ws;
    size_t off = 0;
    auto take = [&](size_t bytes) -> void* {
        void* p = ws + off;
        off += (bytes + 255) & ~(size_t)255;
        return p;
    };
    __hip_bfloat16* enc_bf     = (__hip_bfloat16*)take((size_t)BB * NN * EE * 2);
    float*          att1       = (float*)take((size_t)BB * NN * AA * 4);
    __hip_bfloat16* mean_bf    = (__hip_bfloat16*)take((size_t)BB * EE * 2);
    __hip_bfloat16* h_bf       = (__hip_bfloat16*)take((size_t)BB * DD * 2);
    float*          c_f        = (float*)take((size_t)BB * DD * 4);
    float*          att2       = (float*)take((size_t)BB * AA * 4);
    float*          gate       = (float*)take((size_t)BB * EE * 4);
    __hip_bfloat16* x_bf       = (__hip_bfloat16*)take((size_t)BB * KCAT * 2);
    float*          z          = (float*)take((size_t)BB * 4 * DD * 4);
    float*          preds      = (float*)take((size_t)BB * VPAD * 4);
    __hip_bfloat16* inp_bf     = (__hip_bfloat16*)take((size_t)BB * EMB * 2);
    __hip_bfloat16* Wt_enc_att = (__hip_bfloat16*)take((size_t)AA * EE * 2);
    __hip_bfloat16* Wt_dec_att = (__hip_bfloat16*)take((size_t)AA * DD * 2);
    __hip_bfloat16* Wt_init_h  = (__hip_bfloat16*)take((size_t)DD * EE * 2);
    __hip_bfloat16* Wt_init_c  = (__hip_bfloat16*)take((size_t)DD * EE * 2);
    __hip_bfloat16* Wt_fbeta   = (__hip_bfloat16*)take((size_t)EE * DD * 2);
    __hip_bfloat16* Wt_cat     = (__hip_bfloat16*)take((size_t)(4 * DD) * KCAT * 2);
    __hip_bfloat16* Wt_fc      = (__hip_bfloat16*)take((size_t)VPAD * DD * 2);
    (void)ws_size; (void)n_in; (void)in_sizes; (void)out_size;

    auto cdiv = [](size_t a, size_t b) { return (unsigned)((a + b - 1) / b); };

    // -------- one-time conversions (redone every call; deterministic) --------
    {
        size_t n = (size_t)BB * NN * EE;
        f2bf_kernel<<<cdiv(n, 256), 256, 0, stream>>>(features, enc_bf, n);
    }
    transpose_f2bf_kernel<<<cdiv((size_t)AA * EE, 256), 256, 0, stream>>>(W_enc_att, Wt_enc_att, EE, AA, AA);
    transpose_f2bf_kernel<<<cdiv((size_t)AA * DD, 256), 256, 0, stream>>>(W_dec_att, Wt_dec_att, DD, AA, AA);
    transpose_f2bf_kernel<<<cdiv((size_t)DD * EE, 256), 256, 0, stream>>>(W_init_h, Wt_init_h, EE, DD, DD);
    transpose_f2bf_kernel<<<cdiv((size_t)DD * EE, 256), 256, 0, stream>>>(W_init_c, Wt_init_c, EE, DD, DD);
    transpose_f2bf_kernel<<<cdiv((size_t)EE * DD, 256), 256, 0, stream>>>(W_fbeta, Wt_fbeta, DD, EE, EE);
    transpose_f2bf_kernel<<<cdiv((size_t)VPAD * DD, 256), 256, 0, stream>>>(W_fc, Wt_fc, DD, VV, VPAD);
    build_wcat_kernel<<<cdiv((size_t)(4 * DD) * KCAT, 256), 256, 0, stream>>>(W_ih, W_hh, Wt_cat);
    mean_enc_kernel<<<cdiv((size_t)BB * EE, 256), 256, 0, stream>>>(features, mean_bf);
    init_inp_kernel<<<cdiv((size_t)BB * EMB, 256), 256, 0, stream>>>(embedding, inp_bf);

    // -------- init h, c and precompute att1 --------
    // h0 = mean @ W_init_h + b  (bf16 out only)
    gemm_bf16_kernel<0><<<dim3(DD / 64, BB / 64), 256, 0, stream>>>(
        mean_bf, Wt_init_h, b_init_h, DD, nullptr, h_bf, BB, DD, EE);
    // c0 = mean @ W_init_c + b  (f32 out)
    gemm_bf16_kernel<0><<<dim3(DD / 64, BB / 64), 256, 0, stream>>>(
        mean_bf, Wt_init_c, b_init_c, DD, c_f, nullptr, BB, DD, EE);
    // att1 = enc @ W_enc_att   [12544, 512]
    gemm_bf16_kernel<0><<<dim3(AA / 64, (BB * NN) / 64), 256, 0, stream>>>(
        enc_bf, Wt_enc_att, nullptr, 0, att1, nullptr, BB * NN, AA, EE);

    // -------- 20 decode steps --------
    for (int t = 0; t < TT; ++t) {
        // att2 = h @ W_dec_att
        gemm_bf16_kernel<0><<<dim3(AA / 64, BB / 64), 256, 0, stream>>>(
            h_bf, Wt_dec_att, nullptr, 0, att2, nullptr, BB, AA, DD);
        // gate = sigmoid(h @ W_fbeta + b_fbeta)
        gemm_bf16_kernel<1><<<dim3(EE / 64, BB / 64), 256, 0, stream>>>(
            h_bf, Wt_fbeta, b_fbeta, EE, gate, nullptr, BB, EE, DD);
        // attention -> x[:, 512:2560] = bf16(gate * awe)
        attention_kernel<<<BB, 256, 0, stream>>>(att1, att2, w_full, features, gate, x_bf);
        // x[:, 0:512] = inp ; x[:, 2560:3072] = h
        build_x_kernel<<<cdiv((size_t)BB * (EMB + DD), 256), 256, 0, stream>>>(inp_bf, h_bf, x_bf);
        // z = x @ [W_ih; W_hh] + b_lstm
        gemm_bf16_kernel<0><<<dim3((4 * DD) / 64, BB / 64), 256, 0, stream>>>(
            x_bf, Wt_cat, b_lstm, 4 * DD, z, nullptr, BB, 4 * DD, KCAT);
        // LSTM pointwise -> c, h
        lstm_kernel<<<cdiv((size_t)BB * DD, 256), 256, 0, stream>>>(z, c_f, h_bf);
        // preds = h @ W_fc + b_fc   (padded to VPAD cols)
        gemm_bf16_kernel<0><<<dim3(VPAD / 64, BB / 64), 256, 0, stream>>>(
            h_bf, Wt_fc, b_fc, VV, preds, nullptr, BB, VPAD, DD);
        // argmax + next-token embedding
        argmax_embed_kernel<<<BB, 256, 0, stream>>>(preds, embedding, inp_bf, ids_out, t);
    }
}
